// GraphTransformer_23742579212691
// MI455X (gfx1250) — matrix operand order repaired and checked by reading, not compiled, not run on hardware
//
#include <hip/hip_runtime.h>
#include <hip/hip_bf16.h>

// ---------------------------------------------------------------------------
// Types / WMMA helpers (CDNA5 / gfx1250, wave32)
// ---------------------------------------------------------------------------
typedef __attribute__((ext_vector_type(16))) _Float16 v16h;
typedef __attribute__((ext_vector_type(8)))  _Float16 v8h;
typedef __attribute__((ext_vector_type(4)))  _Float16 v4h;
typedef __attribute__((ext_vector_type(8)))  float    v8f;

__device__ __forceinline__ v8f wmma_f16(v16h a, v16h b, v8f c) {
  // D = A(16x32 f16) * B(32x16 f16) + C(16x16 f32)
  return __builtin_amdgcn_wmma_f32_16x16x32_f16(
      /*neg_a=*/false, a, /*neg_b=*/false, b,
      /*c_mod=*/(short)0, c, /*reuse_a=*/false, /*reuse_b=*/false);
}

__device__ __forceinline__ v8f zero_v8f() {
  v8f v;
#pragma unroll
  for (int i = 0; i < 8; i++) v[i] = 0.0f;
  return v;
}
__device__ __forceinline__ v16h zero_v16h() {
  v16h v;
#pragma unroll
  for (int i = 0; i < 16; i++) v[i] = (_Float16)0;
  return v;
}
__device__ __forceinline__ v8h zero_v8h() {
  v8h v;
#pragma unroll
  for (int i = 0; i < 8; i++) v[i] = (_Float16)0;
  return v;
}

union AFrag {            // 16-bit A-matrix fragment: two 8-half chunks per lane
  v16h v;
  v8h  h[2];
};

// 16-lane group reductions (stay inside each half of the wave; xor bits 0..3)
__device__ __forceinline__ float redmax16(float v) {
#pragma unroll
  for (int off = 1; off < 16; off <<= 1) v = fmaxf(v, __shfl_xor(v, off, 32));
  return v;
}
__device__ __forceinline__ float redsum16(float v) {
#pragma unroll
  for (int off = 1; off < 16; off <<= 1) v += __shfl_xor(v, off, 32);
  return v;
}

// ---------------------------------------------------------------------------
// Problem constants
// ---------------------------------------------------------------------------
#define LAY 3
#define HID 128
#define NHD 8
#define HDM 16
#define FFD 512
#define BSZ 32
#define SEQ 512
#define NTOK (BSZ * SEQ)   // 16384
#define NBINS 50

// ---------------------------------------------------------------------------
// Small elementwise kernels
// ---------------------------------------------------------------------------
__global__ void copy4_kernel(const float* __restrict__ src, float* __restrict__ dst, int n4) {
  int i = blockIdx.x * blockDim.x + threadIdx.x;
  if (i < n4) ((float4*)dst)[i] = ((const float4*)src)[i];
}

__global__ void bins_kernel(const float* __restrict__ dist, unsigned char* __restrict__ bins, int n) {
  int i = blockIdx.x * blockDim.x + threadIdx.x;
  if (i >= n) return;
  int b = (int)(dist[i] * 10.0f);
  b = b < 0 ? 0 : (b > NBINS - 1 ? NBINS - 1 : b);
  bins[i] = (unsigned char)b;
}

// src: [L][K][Nw] f32 (fan_in major)  ->  dst: [L][rowOff+n][k] f16 (transposed)
__global__ void wconv_kernel(const float* __restrict__ src, _Float16* __restrict__ dst,
                             int K, int Nw, int layerStride, int rowOff) {
  long long idx = (long long)blockIdx.x * blockDim.x + threadIdx.x;
  long long total = (long long)LAY * K * Nw;
  if (idx >= total) return;
  int l = (int)(idx / ((long long)K * Nw));
  int rem = (int)(idx % ((long long)K * Nw));
  int k = rem / Nw, n = rem % Nw;
  dst[(size_t)l * layerStride + (size_t)(rowOff + n) * K + k] = (_Float16)src[idx];
}

// one wave per token: LayerNorm(h[128]) -> f16
__global__ __launch_bounds__(256) void ln_kernel(const float* __restrict__ h,
                                                 const float* __restrict__ g,
                                                 const float* __restrict__ be,
                                                 _Float16* __restrict__ out) {
  int w = threadIdx.x >> 5, lane = threadIdx.x & 31;
  int tok = blockIdx.x * 8 + w;
  const float4 xv = *(const float4*)(h + (size_t)tok * HID + lane * 4);
  float xa[4] = {xv.x, xv.y, xv.z, xv.w};
  float s = xa[0] + xa[1] + xa[2] + xa[3];
  float ss = xa[0] * xa[0] + xa[1] * xa[1] + xa[2] * xa[2] + xa[3] * xa[3];
#pragma unroll
  for (int off = 1; off < 32; off <<= 1) {
    s += __shfl_xor(s, off, 32);
    ss += __shfl_xor(ss, off, 32);
  }
  float mu = s * (1.0f / HID);
  float var = ss * (1.0f / HID) - mu * mu;
  float rs = rsqrtf(var + 1e-5f);
  v4h o;
#pragma unroll
  for (int i = 0; i < 4; i++) {
    int c = lane * 4 + i;
    o[i] = (_Float16)((xa[i] - mu) * rs * g[c] + be[c]);
  }
  *(v4h*)(out + (size_t)tok * HID + lane * 4) = o;
}

// GLU: glu[m][j] = u[m][j] * sigmoid(u[m][512+j])
__global__ void glu_kernel(const _Float16* __restrict__ u, _Float16* __restrict__ glu, int n) {
  int i = blockIdx.x * blockDim.x + threadIdx.x;
  if (i >= n) return;
  int m = i >> 9, j = i & 511;
  float a = (float)u[(size_t)m * 1024 + j];
  float b = (float)u[(size_t)m * 1024 + 512 + j];
  glu[i] = (_Float16)(a / (1.0f + __expf(-b)));
}

// ---------------------------------------------------------------------------
// Generic WMMA GEMM (compile-time shape):
//   C[M,NN] = A[M,KK] (f16) * Bt[NN,KK]^T (f16) + bias
// block = 256 thr (8 waves); block tile 128(M) x 64(N); wave tile 16 x 64.
// K-loop: fully unrolled in 128-wide chunks (4x4 v_wmma per chunk), with a
// global_prefetch of the next A chunk for KK > 128.
// MODE 0: out f16 [M,NN]      (FF1)
// MODE 1: resid f32 [M,NN] += (O-proj / FF2, NN==128)
// MODE 2: QKV scatter (NN==384): Q,K row-major [M,128] f16; V transposed
//         Vt[b][hd][d][tok] f16
// ---------------------------------------------------------------------------
template <int MODE, int NN, int KK>
__global__ __launch_bounds__(256) void gemm_kernel(
    const _Float16* __restrict__ A, const _Float16* __restrict__ Bt,
    const float* __restrict__ bias0, const float* __restrict__ bias1,
    const float* __restrict__ bias2, float* __restrict__ resid,
    _Float16* __restrict__ O0, _Float16* __restrict__ O1,
    _Float16* __restrict__ O2) {
  const int tid = threadIdx.x;
  const int w = tid >> 5, lane = tid & 31;
  const int l15 = lane & 15, kh = lane >> 4;
  const int m0 = blockIdx.y * 128 + w * 16;
  const int n0 = blockIdx.x * 64;
  const _Float16* Arow = A + (size_t)(m0 + l15) * KK;
  const _Float16* Bbase = Bt + (size_t)(n0 + l15) * KK;

  v8f acc[4];
#pragma unroll
  for (int j = 0; j < 4; j++) acc[j] = zero_v8f();

#pragma unroll
  for (int kc = 0; kc < KK; kc += 128) {
    if (kc + 128 < KK)  // prefetch next A chunk (global_prefetch_b8)
      __builtin_prefetch(Arow + kc + 128, 0, 0);
#pragma unroll
    for (int ki = 0; ki < 4; ki++) {
      const int k0 = kc + ki * 32;
      AFrag a;
      a.h[0] = *(const v8h*)(Arow + k0 + 16 * kh);
      a.h[1] = *(const v8h*)(Arow + k0 + 16 * kh + 8);
      const _Float16* Bp = Bbase + k0 + 16 * kh;
      v16h b0 = *(const v16h*)(Bp);
      v16h b1 = *(const v16h*)(Bp + (size_t)16 * KK);
      v16h b2 = *(const v16h*)(Bp + (size_t)32 * KK);
      v16h b3 = *(const v16h*)(Bp + (size_t)48 * KK);
      acc[0] = wmma_f16(a.v, b0, acc[0]);
      acc[1] = wmma_f16(a.v, b1, acc[1]);
      acc[2] = wmma_f16(a.v, b2, acc[2]);
      acc[3] = wmma_f16(a.v, b3, acc[3]);
    }
  }

#pragma unroll
  for (int j = 0; j < 4; j++) {
    const int n = n0 + 16 * j + l15;
#pragma unroll
    for (int r = 0; r < 8; r++) {
      const int m = m0 + r + 8 * kh;
      float val = acc[j][r];
      if (MODE == 0) {
        O0[(size_t)m * NN + n] = (_Float16)(val + bias0[n]);
      } else if (MODE == 1) {
        resid[(size_t)m * NN + n] += val + bias0[n];
      } else {  // MODE 2: QKV
        if (n < 128) {
          O0[(size_t)m * HID + n] = (_Float16)(val + bias0[n]);
        } else if (n < 256) {
          const int nn = n - 128;
          O1[(size_t)m * HID + nn] = (_Float16)(val + bias1[nn]);
        } else {
          const int v = n - 256;
          const int hh = v >> 4, dd = v & 15;
          const int bb = m >> 9, tok = m & 511;
          O2[((((size_t)bb * NHD + hh) * HDM + dd) << 9) + tok] =
              (_Float16)(val + bias2[v]);
        }
      }
    }
  }
}

// ---------------------------------------------------------------------------
// Flash attention: one wave per 16-row q-tile for fixed (b, head).
// scores via v_wmma (HD=16 zero-padded to K=32); online softmax with 16-lane
// shuffle reductions; P re-laid-out C->A through 1KB wave-private LDS; P@V
// accumulated with one v_wmma per 32 keys (N==HD==16 exactly).
// grid: B*NH*4 blocks, 256 thr (8 waves, one q-tile each).
// ---------------------------------------------------------------------------
__global__ __launch_bounds__(256) void attn_kernel(
    const _Float16* __restrict__ Qh, const _Float16* __restrict__ Kh,
    const _Float16* __restrict__ Vt, const unsigned char* __restrict__ bins,
    const unsigned char* __restrict__ mask, const float* __restrict__ emb_l,
    const float* __restrict__ abias_l, _Float16* __restrict__ attn_out) {
  __shared__ float emb_s[64];
  __shared__ unsigned char mask_s[SEQ];
  __shared__ __align__(32) _Float16 P[8][16][32];

  const int tid = threadIdx.x;
  const int w = tid >> 5, lane = tid & 31;
  const int l15 = lane & 15, kh = lane >> 4;

  const int qgroup = blockIdx.x & 3;
  const int hd = (blockIdx.x >> 2) & 7;
  const int b = blockIdx.x >> 5;
  const int q0 = (qgroup * 8 + w) * 16;

  if (tid < NBINS) emb_s[tid] = emb_l[tid * NHD + hd];
  mask_s[tid] = mask[(size_t)b * SEQ + tid];
  mask_s[tid + 256] = mask[(size_t)b * SEQ + tid + 256];
  __syncthreads();

  const float ab = abias_l[hd];

  // Q fragment (K-dim = head dim 16, padded to 32 with zeros)
  AFrag qa;
  if (kh == 0)
    qa.v = *(const v16h*)(Qh + (size_t)(b * SEQ + q0 + l15) * HID + hd * HDM);
  else
    qa.v = zero_v16h();

  float rmax[8], rsum[8], cf[8];
#pragma unroll
  for (int r = 0; r < 8; r++) {
    rmax[r] = -1e30f;
    rsum[r] = 0.0f;
  }
  v8f o = zero_v8f();

  const unsigned char* binsb = bins + (size_t)b * SEQ * SEQ;

  for (int kb = 0; kb < SEQ; kb += 32) {
    v8f S[2];
#pragma unroll
    for (int s = 0; s < 2; s++) {
      const int kbs = kb + 16 * s;
      v16h kf;
      if (kh == 0)
        kf = *(const v16h*)(Kh + (size_t)(b * SEQ + kbs + l15) * HID + hd * HDM);
      else
        kf = zero_v16h();
      S[s] = wmma_f16(qa.v, kf, zero_v8f());
      const int key = kbs + l15;
      const bool km = (mask_s[key] != 0);
#pragma unroll
      for (int r = 0; r < 8; r++) {
        const int q = q0 + r + 8 * kh;
        float sc = S[s][r] * 0.25f + emb_s[binsb[(size_t)q * SEQ + key]] + ab;
        S[s][r] = km ? -1e9f : sc;
      }
    }
    // online softmax per row (rows live across 16-lane halves)
#pragma unroll
    for (int r = 0; r < 8; r++) {
      float tm = redmax16(fmaxf(S[0][r], S[1][r]));
      float nm = fmaxf(rmax[r], tm);
      cf[r] = __expf(rmax[r] - nm);
      rmax[r] = nm;
      float p0 = __expf(S[0][r] - nm);
      float p1 = __expf(S[1][r] - nm);
      rsum[r] = rsum[r] * cf[r] + redsum16(p0 + p1);
      P[w][r + 8 * kh][l15] = (_Float16)p0;
      P[w][r + 8 * kh][16 + l15] = (_Float16)p1;
    }
    asm volatile("s_wait_dscnt 0" ::: "memory");  // DS in-order per wave
    AFrag pa;
    pa.h[0] = *(const v8h*)(&P[w][l15][16 * kh]);
    pa.h[1] = *(const v8h*)(&P[w][l15][16 * kh + 8]);
    v16h vf = *(const v16h*)(Vt + (((size_t)(b * NHD + hd) * HDM + l15) << 9) + kb + 16 * kh);
#pragma unroll
    for (int r = 0; r < 8; r++) o[r] *= cf[r];
    o = wmma_f16(pa.v, vf, o);
  }

#pragma unroll
  for (int r = 0; r < 8; r++) {
    attn_out[(size_t)(b * SEQ + q0 + r + 8 * kh) * HID + hd * HDM + l15] =
        (_Float16)(o[r] / rsum[r]);
  }
}

// ---------------------------------------------------------------------------
// Host launcher
// ---------------------------------------------------------------------------
extern "C" void kernel_launch(void* const* d_in, const int* in_sizes, int n_in,
                              void* d_out, int out_size, void* d_ws, size_t ws_size,
                              hipStream_t stream) {
  (void)in_sizes; (void)n_in; (void)out_size; (void)ws_size;
  const float* x      = (const float*)d_in[0];
  const float* dist   = (const float*)d_in[1];
  const unsigned char* mask = (const unsigned char*)d_in[2];
  const float* Wq = (const float*)d_in[3];  const float* bq = (const float*)d_in[4];
  const float* Wk = (const float*)d_in[5];  const float* bk = (const float*)d_in[6];
  const float* Wv = (const float*)d_in[7];  const float* bv = (const float*)d_in[8];
  const float* Wo = (const float*)d_in[9];  const float* bo = (const float*)d_in[10];
  const float* dist_emb  = (const float*)d_in[11];
  const float* attn_bias = (const float*)d_in[12];
  const float* g1 = (const float*)d_in[13]; const float* b1 = (const float*)d_in[14];
  const float* g2 = (const float*)d_in[15]; const float* b2 = (const float*)d_in[16];
  const float* Wf1 = (const float*)d_in[17]; const float* bf1 = (const float*)d_in[18];
  const float* Wf2 = (const float*)d_in[19]; const float* bf2 = (const float*)d_in[20];

  float* h = (float*)d_out;  // running residual stream [B,N,H] f32
  char* ws = (char*)d_ws;

  // workspace layout (bytes)
  unsigned char* bins = (unsigned char*)(ws + 0);            // 8 MB
  _Float16* Qh   = (_Float16*)(ws + 8388608);                // 4 MB [M,128]
  _Float16* Khb  = (_Float16*)(ws + 12582912);               // 4 MB [M,128]
  _Float16* Vt   = (_Float16*)(ws + 16777216);               // 4 MB [B,NH,HD,N]
  _Float16* ln   = (_Float16*)(ws + 20971520);               // 4 MB [M,128]
  _Float16* ao   = (_Float16*)(ws + 25165824);               // 4 MB [M,128]
  _Float16* u    = (_Float16*)(ws + 29360128);               // 32 MB [M,1024]
  _Float16* glu  = (_Float16*)(ws + 62914560);               // 16 MB [M,512]
  _Float16* Wqkvt = (_Float16*)(ws + 79691776);              // [L][384][128]
  _Float16* Wot   = (_Float16*)(ws + 79986688);              // [L][128][128]
  _Float16* Wf1t  = (_Float16*)(ws + 80084992);              // [L][1024][128]
  _Float16* Wf2t  = (_Float16*)(ws + 80871424);              // [L][128][512]

  // init: h = x; bins = clip(int(dist*10),0,49)
  copy4_kernel<<<(NTOK * HID / 4 + 255) / 256, 256, 0, stream>>>(x, h, NTOK * HID / 4);
  bins_kernel<<<(BSZ * SEQ * SEQ + 255) / 256, 256, 0, stream>>>(dist, bins, BSZ * SEQ * SEQ);

  // weight conversion (transposed f16)
  {
    int t = LAY * 128 * 128;
    wconv_kernel<<<(t + 255) / 256, 256, 0, stream>>>(Wq, Wqkvt, 128, 128, 384 * 128, 0);
    wconv_kernel<<<(t + 255) / 256, 256, 0, stream>>>(Wk, Wqkvt, 128, 128, 384 * 128, 128);
    wconv_kernel<<<(t + 255) / 256, 256, 0, stream>>>(Wv, Wqkvt, 128, 128, 384 * 128, 256);
    wconv_kernel<<<(t + 255) / 256, 256, 0, stream>>>(Wo, Wot, 128, 128, 128 * 128, 0);
    int t1 = LAY * 128 * 1024;
    wconv_kernel<<<(t1 + 255) / 256, 256, 0, stream>>>(Wf1, Wf1t, 128, 1024, 1024 * 128, 0);
    int t2 = LAY * 512 * 128;
    wconv_kernel<<<(t2 + 255) / 256, 256, 0, stream>>>(Wf2, Wf2t, 512, 128, 128 * 512, 0);
  }

  for (int l = 0; l < LAY; l++) {
    // --- attention block (pre-LN) ---
    ln_kernel<<<NTOK / 8, 256, 0, stream>>>(h, g1 + l * HID, b1 + l * HID, ln);
    gemm_kernel<2, 384, 128><<<dim3(384 / 64, NTOK / 128), 256, 0, stream>>>(
        ln, Wqkvt + (size_t)l * 384 * 128, bq + l * HID, bk + l * HID, bv + l * HID,
        nullptr, Qh, Khb, Vt);
    attn_kernel<<<BSZ * NHD * 4, 256, 0, stream>>>(
        Qh, Khb, Vt, bins, mask, dist_emb + (size_t)l * NBINS * NHD,
        attn_bias + (size_t)l * NHD, ao);
    gemm_kernel<1, 128, 128><<<dim3(128 / 64, NTOK / 128), 256, 0, stream>>>(
        ao, Wot + (size_t)l * 128 * 128, bo + l * HID, nullptr, nullptr,
        h, nullptr, nullptr, nullptr);

    // --- GLU feed-forward block (pre-LN) ---
    ln_kernel<<<NTOK / 8, 256, 0, stream>>>(h, g2 + l * HID, b2 + l * HID, ln);
    gemm_kernel<0, 1024, 128><<<dim3(1024 / 64, NTOK / 128), 256, 0, stream>>>(
        ln, Wf1t + (size_t)l * 1024 * 128, bf1 + l * 1024, nullptr, nullptr,
        nullptr, u, nullptr, nullptr);
    glu_kernel<<<(NTOK * FFD + 255) / 256, 256, 0, stream>>>(u, glu, NTOK * FFD);
    gemm_kernel<1, 128, 512><<<dim3(128 / 64, NTOK / 128), 256, 0, stream>>>(
        glu, Wf2t + (size_t)l * 128 * 512, bf2 + l * HID, nullptr, nullptr,
        h, nullptr, nullptr, nullptr);
  }
}
